// VQVAE_80315888435963
// MI455X (gfx1250) — compile-verified, matrix operand order checked by reading, not run on hardware
//
#include <hip/hip_runtime.h>
#include <hip/hip_bf16.h>

typedef __attribute__((ext_vector_type(16))) _Float16 v16h;
typedef __attribute__((ext_vector_type(8)))  float    v8f;

#define BM 64
#define BN 64
#define BK 32
#define AST 40   // A LDS row stride (halves) = 80B (16B multiple, not 256B multiple)
#define BST 72   // B LDS row stride (halves) = 144B (16B multiple, not 256B multiple)

__device__ __forceinline__ float act_apply(float v, int act) {
  if (act == 1) return v > 0.f ? v : 0.01f * v;     // leaky relu
  if (act == 2) return v > 0.f ? v : 0.f;           // relu
  if (act == 3) return 1.f / (1.f + __expf(-v));    // sigmoid
  return v;
}

// Implicit-GEMM conv: out[M=Cout][P=N*OH*OW] = W[M][K=Cin*KH*KW] * Patch[K][P]
// TMODE=0: standard conv (dil=1, generic KH/KW, OIHW weights)
// TMODE=1: ConvTranspose2d as lhs-dilated conv (dil=2, KH=KW=4 hardcoded,
//          weights [Cin,Cout,4,4] read flipped+transposed)
// Requires: OW a multiple of 16 (true for all layers here), so each thread's
// 16-wide B segment lies within a single output row.
template <int TMODE>
__global__ __launch_bounds__(128)
void conv_gemm_wmma(const float* __restrict__ in, const float* __restrict__ w,
                    const float* __restrict__ bias, const float* __restrict__ res,
                    float* __restrict__ out,
                    int Nimg, int Cin, int H, int W, int Cout, int KH, int KW,
                    int OH, int OW, int stride, int pad, int act)
{
  __shared__ _Float16 Al[BM * AST];
  __shared__ _Float16 Bl[BK * BST];

  const int tid  = threadIdx.x;
  const int lane = tid & 31;
  const int wid  = tid >> 5;      // 0..3
  const int waveM = wid >> 1;     // 0..1  (32-row slab)
  const int waveN = wid & 1;      // 0..1  (32-col slab)
  const int ohw  = OH * OW;
  const int khw  = TMODE ? 16 : KH * KW;
  const int KWe  = TMODE ? 4 : KW;
  const int Kdim = Cin * khw;
  const int Ptot = Nimg * ohw;
  const int p0 = blockIdx.x * BN;
  const int m0 = blockIdx.y * BM;

  __builtin_prefetch(w + (size_t)m0 * (size_t)Kdim, 0, 1);

  // ---- per-thread invariants for staging (hoisted out of the K loop) ----
  // B tile: each thread owns one K-row (kkT) and 16 consecutive columns,
  // which all fall in one output row of one image.
  const int kkT = tid >> 2;             // 0..31
  const int nnB = (tid & 3) * 16;       // 0,16,32,48
  const int Pseg = p0 + nnB;
  const bool pvSeg = (Pseg + 15) < Ptot;
  const int nI  = Pseg / ohw, psp = Pseg - nI * ohw;
  const int oh  = psp / OW,   owB = psp - oh * OW;
  const int inBase = nI * Cin * H * W;
  const int ohS0 = oh * stride - pad;   // ihe = ohS0 + kh (row constant per segment)
  const int owS0 = owB * stride - pad;  // iwe_i = owS0 + i*stride + kw
  // A tile: each thread owns one M-row (rowA) and 16 consecutive K columns.
  const int rowA = tid >> 1;            // 0..63
  const int colB = (tid & 1) * 16;      // 0 or 16
  const int mA   = m0 + rowA;
  const float* wRow = w + (size_t)(mA < Cout ? mA : 0) * Kdim;

  v8f acc[2][2] = {};

  for (int k0 = 0; k0 < Kdim; k0 += BK) {
    // ---- stage A tile (weights, Cout x K) as f16 ----
    if (TMODE == 0) {
      const float* wp = wRow + k0 + colB;
      if (mA < Cout && (k0 + colB + 16) <= Kdim) {
        #pragma unroll
        for (int i = 0; i < 16; ++i)
          Al[rowA * AST + colB + i] = (_Float16)wp[i];
      } else {
        #pragma unroll
        for (int i = 0; i < 16; ++i) {
          float v = (mA < Cout && (k0 + colB + i) < Kdim) ? wp[i] : 0.f;
          Al[rowA * AST + colB + i] = (_Float16)v;
        }
      }
    } else {
      #pragma unroll
      for (int i = 0; i < 16; ++i) {
        int k = k0 + colB + i;
        float v = 0.f;
        if (mA < Cout && k < Kdim) {
          int ic = k >> 4, r2 = k & 15;
          int kh = r2 >> 2, kw = r2 & 3;
          v = w[((size_t)(ic * Cout + mA) * 4 + (3 - kh)) * 4 + (3 - kw)];
        }
        Al[rowA * AST + colB + i] = (_Float16)v;
      }
    }
    // ---- stage B tile (im2col patches, K x P) as f16 ----
    {
      int k = k0 + kkT;
      bool kva = k < Kdim;
      int ic, kh, kw;
      if (TMODE == 0) { ic = k / khw; int r2 = k - ic * khw; kh = r2 / KWe; kw = r2 - kh * KWe; }
      else            { ic = k >> 4;  int r2 = k & 15;       kh = r2 >> 2;  kw = r2 & 3; }
      const int ihe = ohS0 + kh;
      const int rowOk = kva && pvSeg && ihe >= 0;
      if (TMODE == 0) {
        const int iwe0 = owS0 + kw;
        if (stride == 1 && rowOk && ihe < H && iwe0 >= 0 && (iwe0 + 15) < W) {
          // interior fast path: 16 contiguous unguarded loads (merges to b128)
          const float* src = in + inBase + ic * H * W + ihe * W + iwe0;
          #pragma unroll
          for (int i = 0; i < 16; ++i)
            Bl[kkT * BST + nnB + i] = (_Float16)src[i];
        } else {
          const bool hok = rowOk && ihe < H;
          const int rowBase = inBase + ic * H * W + ihe * W;
          #pragma unroll
          for (int i = 0; i < 16; ++i) {
            int iwe = iwe0 + i * stride;
            float v = (hok && iwe >= 0 && iwe < W) ? in[rowBase + iwe] : 0.f;
            Bl[kkT * BST + nnB + i] = (_Float16)v;
          }
        }
      } else {
        const bool hok = rowOk && !(ihe & 1) && (ihe >> 1) < H;
        const int rowBase = inBase + ic * H * W + (ihe >> 1) * W;
        #pragma unroll
        for (int i = 0; i < 16; ++i) {
          int iwe = owS0 + i + kw;       // stride==1 in dilated space
          float v = 0.f;
          if (hok && iwe >= 0 && !(iwe & 1) && (iwe >> 1) < W)
            v = in[rowBase + (iwe >> 1)];
          Bl[kkT * BST + nnB + i] = (_Float16)v;
        }
      }
    }
    __syncthreads();

    // ---- fragments per the CDNA5 16-bit A layout: k = (j/8)*16 + (lane/16)*8 + (j%8)
    v16h af[2], bf[2];
    #pragma unroll
    for (int t = 0; t < 2; ++t) {
      int row = waveM * 32 + t * 16 + (lane & 15);
      #pragma unroll
      for (int j = 0; j < 16; ++j) {
        int col = ((j >> 3) << 4) + ((lane >> 4) << 3) + (j & 7);
        af[t][j] = Al[row * AST + col];
      }
    }
    #pragma unroll
    for (int u = 0; u < 2; ++u) {
      int coln = waveN * 32 + u * 16 + (lane & 15);
      #pragma unroll
      for (int j = 0; j < 16; ++j) {
        int kk = ((j >> 3) << 4) + ((lane >> 4) << 3) + (j & 7);
        bf[u][j] = Bl[kk * BST + coln];
      }
    }
    #pragma unroll
    for (int t = 0; t < 2; ++t)
      #pragma unroll
      for (int u = 0; u < 2; ++u)
        acc[t][u] = __builtin_amdgcn_wmma_f32_16x16x32_f16(
            false, af[t], false, bf[u], (short)0, acc[t][u], false, false);
    __syncthreads();
  }

  // ---- epilogue: bias + optional residual + activation, NCHW scatter ----
  #pragma unroll
  for (int u = 0; u < 2; ++u) {
    int P = p0 + waveN * 32 + u * 16 + (lane & 15);
    if (P < Ptot) {
      int nIe = P / ohw, pspE = P - nIe * ohw;
      size_t base = (size_t)nIe * Cout * ohw + pspE;
      #pragma unroll
      for (int t = 0; t < 2; ++t)
        #pragma unroll
        for (int r = 0; r < 8; ++r) {
          int oc = m0 + waveM * 32 + t * 16 + r + ((lane >> 4) << 3);
          if (oc < Cout) {
            size_t o = base + (size_t)oc * ohw;
            float v = acc[t][u][r] + bias[oc];
            if (res) v += res[o];
            out[o] = act_apply(v, act);
          }
        }
    }
  }
}

// ---- VQ: distances via WMMA (16 pos x 16 codes x K64), argmin, gather, loss ----
__global__ __launch_bounds__(128)
void vq_wmma(const float* __restrict__ z, const float* __restrict__ cb,
             const float* __restrict__ cnorm, float* __restrict__ q,
             float* __restrict__ accum, int HW)
{
  __shared__ _Float16 Zl[4][16 * 68];
  __shared__ int idxl[4][16];
  const int tid = threadIdx.x, lane = tid & 31, wid = tid >> 5;
  const int P0 = (blockIdx.x * 4 + wid) * 16;
  const int D = 64;

  // stage 16x64 latent tile per wave, f32 -> f16 (NCHW gather)
  for (int i = 0; i < 32; ++i) {
    int e = lane * 32 + i;
    int row = e >> 6, d = e & 63;
    int P = P0 + row;
    int nI = P / HW, psp = P - nI * HW;
    Zl[wid][row * 68 + d] = (_Float16)z[(size_t)(nI * D + d) * HW + psp];
  }
  __syncthreads();

  v16h af[2];
  #pragma unroll
  for (int s = 0; s < 2; ++s) {
    int row = lane & 15;
    #pragma unroll
    for (int j = 0; j < 16; ++j) {
      int col = s * 32 + ((j >> 3) << 4) + ((lane >> 4) << 3) + (j & 7);
      af[s][j] = Zl[wid][row * 68 + col];
    }
  }

  float best[8]; int bidx[8];
  #pragma unroll
  for (int r = 0; r < 8; ++r) { best[r] = 3.0e38f; bidx[r] = 0; }

  for (int tt = 0; tt < 32; ++tt) {       // 512 codes / 16
    v8f acc = {};
    int code = tt * 16 + (lane & 15);
    #pragma unroll
    for (int s = 0; s < 2; ++s) {
      v16h bf;
      #pragma unroll
      for (int j = 0; j < 16; ++j) {
        int k = s * 32 + ((j >> 3) << 4) + ((lane >> 4) << 3) + (j & 7);
        bf[j] = (_Float16)cb[code * 64 + k];
      }
      acc = __builtin_amdgcn_wmma_f32_16x16x32_f16(
          false, af[s], false, bf, (short)0, acc, false, false);
    }
    float cn = cnorm[code];
    #pragma unroll
    for (int r = 0; r < 8; ++r) {
      float sc = cn - 2.f * acc[r];       // ||f||^2 constant per row, irrelevant for argmin
      if (sc < best[r]) { best[r] = sc; bidx[r] = code; }
    }
  }
  // reduce argmin across the 16 lanes holding the N dimension
  #pragma unroll
  for (int m = 1; m < 16; m <<= 1)
    #pragma unroll
    for (int r = 0; r < 8; ++r) {
      float ob = __shfl_xor(best[r], m, 32);
      int   oi = __shfl_xor(bidx[r], m, 32);
      if (ob < best[r] || (ob == best[r] && oi < bidx[r])) { best[r] = ob; bidx[r] = oi; }
    }
  if ((lane & 15) == 0) {
    #pragma unroll
    for (int r = 0; r < 8; ++r)
      idxl[wid][r + ((lane >> 4) << 3)] = bidx[r];
  }
  __syncthreads();

  // gather quantized vectors (straight-through forward == codebook[idx]) + loss
  float lsum = 0.f;
  for (int i = 0; i < 32; ++i) {
    int e = lane * 32 + i;
    int row = e >> 6, d = e & 63;
    int P = P0 + row;
    int nI = P / HW, psp = P - nI * HW;
    size_t o = (size_t)(nI * D + d) * HW + psp;
    float qv = cb[idxl[wid][row] * 64 + d];
    float lv = z[o];
    q[o] = qv;
    float df = qv - lv;
    lsum += df * df;
  }
  #pragma unroll
  for (int m = 1; m < 32; m <<= 1) lsum += __shfl_xor(lsum, m, 32);
  if (lane == 0) atomicAdd(accum, lsum);
}

__global__ void cnorm_k(const float* __restrict__ cb, float* __restrict__ cn) {
  int c = blockIdx.x * blockDim.x + threadIdx.x;
  if (c < 512) {
    float s = 0.f;
    for (int d = 0; d < 64; ++d) { float v = cb[c * 64 + d]; s += v * v; }
    cn[c] = s;
  }
}
__global__ void zero_k(float* p) { if (threadIdx.x == 0 && blockIdx.x == 0) p[0] = 0.f; }
__global__ void loss_k(const float* a, float* o) {
  if (threadIdx.x == 0 && blockIdx.x == 0)
    o[0] = a[0] * 1.25f / (131072.f * 64.f);   // (BETA+1)*mean, commit==embed fwd value
}

static inline void conv_std(hipStream_t s, const float* in, const float* w, const float* b,
                            const float* res, float* out, int N, int Cin, int H, int W,
                            int Cout, int K, int OH, int OW, int stride, int pad, int act) {
  int Ptot = N * OH * OW;
  dim3 grid((Ptot + BN - 1) / BN, (Cout + BM - 1) / BM);
  conv_gemm_wmma<0><<<grid, 128, 0, s>>>(in, w, b, res, out, N, Cin, H, W, Cout, K, K,
                                         OH, OW, stride, pad, act);
}
static inline void conv_tr(hipStream_t s, const float* in, const float* w, const float* b,
                           float* out, int N, int Cin, int H, int W, int Cout,
                           int OH, int OW, int act) {
  int Ptot = N * OH * OW;
  dim3 grid((Ptot + BN - 1) / BN, (Cout + BM - 1) / BM);
  conv_gemm_wmma<1><<<grid, 128, 0, s>>>(in, w, b, nullptr, out, N, Cin, H, W, Cout, 4, 4,
                                         OH, OW, 1, 2, act);
}

extern "C" void kernel_launch(void* const* d_in, const int* in_sizes, int n_in,
                              void* d_out, int out_size, void* d_ws, size_t ws_size,
                              hipStream_t stream) {
  const float* x      = (const float*)d_in[0];
  const float* enc_w1 = (const float*)d_in[1];  const float* enc_b1 = (const float*)d_in[2];
  const float* enc_w2 = (const float*)d_in[3];  const float* enc_b2 = (const float*)d_in[4];
  const float* er1_w1 = (const float*)d_in[5];  const float* er1_b1 = (const float*)d_in[6];
  const float* er1_w2 = (const float*)d_in[7];  const float* er1_b2 = (const float*)d_in[8];
  const float* er2_w1 = (const float*)d_in[9];  const float* er2_b1 = (const float*)d_in[10];
  const float* er2_w2 = (const float*)d_in[11]; const float* er2_b2 = (const float*)d_in[12];
  const float* enc_wz = (const float*)d_in[13]; const float* enc_bz = (const float*)d_in[14];
  const float* cb     = (const float*)d_in[15];
  const float* dec_w0 = (const float*)d_in[16]; const float* dec_b0 = (const float*)d_in[17];
  const float* dr1_w1 = (const float*)d_in[18]; const float* dr1_b1 = (const float*)d_in[19];
  const float* dr1_w2 = (const float*)d_in[20]; const float* dr1_b2 = (const float*)d_in[21];
  const float* dr2_w1 = (const float*)d_in[22]; const float* dr2_b1 = (const float*)d_in[23];
  const float* dr2_w2 = (const float*)d_in[24]; const float* dr2_b2 = (const float*)d_in[25];
  const float* dct1_w = (const float*)d_in[26]; const float* dct1_b = (const float*)d_in[27];
  const float* dct2_w = (const float*)d_in[28]; const float* dct2_b = (const float*)d_in[29];
  float* out = (float*)d_out;

  // workspace layout (floats)
  float* A1 = (float*)d_ws;                    // 32*64*128*128 = 33,554,432
  float* A2 = A1 + 33554432;                   // 32*128*64*64  = 16,777,216
  float* A3 = A2 + 16777216;
  float* T1 = A1;                              // alias: A1 free between conv2 and dct1
  float* Z  = A3 + 16777216;                   // 32*64*64*64 = 8,388,608
  float* Q  = Z + 8388608;
  float* CN = Q + 8388608;                     // 512
  float* AC = CN + 512;                        // 1

  // ---------------- encoder ----------------
  conv_std(stream, x,  enc_w1, enc_b1, nullptr, A1, 32,   1, 256, 256,  64, 4, 128, 128, 2, 1, 1);
  conv_std(stream, A1, enc_w2, enc_b2, nullptr, A2, 32,  64, 128, 128, 128, 4,  64,  64, 2, 1, 1);
  conv_std(stream, A2, er1_w1, er1_b1, nullptr, T1, 32, 128,  64,  64, 128, 3,  64,  64, 1, 1, 2);
  conv_std(stream, T1, er1_w2, er1_b2, A2,      A3, 32, 128,  64,  64, 128, 1,  64,  64, 1, 0, 0);
  conv_std(stream, A3, er2_w1, er2_b1, nullptr, T1, 32, 128,  64,  64, 128, 3,  64,  64, 1, 1, 2);
  conv_std(stream, T1, er2_w2, er2_b2, A3,      A2, 32, 128,  64,  64, 128, 1,  64,  64, 1, 0, 0);
  conv_std(stream, A2, enc_wz, enc_bz, nullptr, Z,  32, 128,  64,  64,  64, 1,  64,  64, 1, 0, 0);

  // ---------------- vector quantize ----------------
  cnorm_k<<<2, 256, 0, stream>>>(cb, CN);
  zero_k<<<1, 32, 0, stream>>>(AC);
  vq_wmma<<<131072 / 64, 128, 0, stream>>>(Z, cb, CN, Q, AC, 4096);

  // ---------------- decoder ----------------
  conv_std(stream, Q,  dec_w0, dec_b0, nullptr, A2, 32,  64,  64,  64, 128, 3,  64,  64, 1, 1, 0);
  conv_std(stream, A2, dr1_w1, dr1_b1, nullptr, T1, 32, 128,  64,  64, 128, 3,  64,  64, 1, 1, 2);
  conv_std(stream, T1, dr1_w2, dr1_b2, A2,      A3, 32, 128,  64,  64, 128, 1,  64,  64, 1, 0, 0);
  conv_std(stream, A3, dr2_w1, dr2_b1, nullptr, T1, 32, 128,  64,  64, 128, 3,  64,  64, 1, 1, 2);
  conv_std(stream, T1, dr2_w2, dr2_b2, A3,      A2, 32, 128,  64,  64, 128, 1,  64,  64, 1, 0, 0);
  // transposed convs as lhs-dilated convs (stride 1, pad = k-1-pad = 2, dil = 2)
  conv_tr(stream, A2, dct1_w, dct1_b, A1, 32, 128,  64,  64,  64, 128, 128, 1);
  conv_tr(stream, A1, dct2_w, dct2_b, out, 32,  64, 128, 128,   1, 256, 256, 3);

  // passthrough x and scalar loss
  hipMemcpyAsync(out + 2097152, x, 2097152 * sizeof(float), hipMemcpyDeviceToDevice, stream);
  loss_k<<<1, 32, 0, stream>>>(AC, out + 4194304);
}
